// RelativeMHSABlock_13864154432324
// MI455X (gfx1250) — matrix operand order repaired and checked by reading, not compiled, not run on hardware
//
#include <hip/hip_runtime.h>
#include <hip/hip_bf16.h>

typedef __bf16 bf16;
typedef __attribute__((ext_vector_type(16))) __bf16 v16bf;
typedef __attribute__((ext_vector_type(8)))  __bf16 v8bf;
typedef __attribute__((ext_vector_type(8)))  float  v8f;
typedef __attribute__((ext_vector_type(4)))  unsigned int u32x4;
typedef __attribute__((ext_vector_type(4)))  int i32x4;
typedef __attribute__((ext_vector_type(8)))  int i32x8;

#define BATCH   32
#define N_TOK   785
#define DIM     768
#define HID     3072
#define NH      12
#define HD      64
#define QKV3    2304
#define M_ROWS  (BATCH * N_TOK)   // 25120
#define NKT     25                // key tiles of 32 (800 padded)
#define NQT     50                // query tiles of 16 (800 padded)

#if __has_builtin(__builtin_amdgcn_tensor_load_to_lds) && __has_builtin(__builtin_amdgcn_s_wait_tensorcnt)
#define HAVE_TDM 1
#else
#define HAVE_TDM 0
#endif

// ---------------- elementwise f32 -> bf16 weight conversion ----------------
__global__ void cvt_bf16_kernel(const float* __restrict__ src, bf16* __restrict__ dst, int n) {
    int i = blockIdx.x * 256 + threadIdx.x;
    if (i < n) dst[i] = (bf16)src[i];
}

// ---------------- relative-position index (H = W = 28, EXTRA = 1) ----------------
__device__ __forceinline__ int rel_idx(int m, int n) {
    if (m == 0 || n == 0) return 55 * 55;                // pad row of table (3025)
    const int p = m - 1, q = n - 1;
    const int di = p / 28 - q / 28 + 27;
    const int dj = p % 28 - q % 28 + 27;
    return di * 55 + dj;
}

// ---------------- precompute bias pre-swizzled into TRANSPOSED C-fragment order ----------------
__global__ __launch_bounds__(512) void bias_kernel(
    const float* __restrict__ rel_table, float* __restrict__ biasS)
{
    const int h  = blockIdx.y;
    const int qt = blockIdx.x / NKT, kt = blockIdx.x % NKT;
    const int t  = threadIdx.x;
    const int lane = t >> 4, idx = t & 15;
    const int hs = idx >> 3, r = idx & 7;
    const int mg = min(qt * 16 + (lane & 15), N_TOK - 1);
    const int ng = kt * 32 + hs * 16 + r + 8 * (lane >> 4);
    const float v = (ng < N_TOK) ? rel_table[(size_t)rel_idx(mg, ng) * NH + h] : -1.0e30f;
    biasS[((size_t)h * (NQT * NKT) + blockIdx.x) * 512 + t] = v;
}

// ---------------- LayerNorm (768) -> bf16 ----------------
__global__ __launch_bounds__(256) void ln_bf16_kernel(
    const float* __restrict__ x, const float* __restrict__ w,
    const float* __restrict__ b, bf16* __restrict__ out)
{
    __shared__ float red[256];
    const int row = blockIdx.x, t = threadIdx.x;
    const float* xr = x + (size_t)row * DIM;
    float v[3], s = 0.f;
#pragma unroll
    for (int i = 0; i < 3; i++) { v[i] = xr[t + 256 * i]; s += v[i]; }
    red[t] = s; __syncthreads();
    for (int o = 128; o > 0; o >>= 1) { if (t < o) red[t] += red[t + o]; __syncthreads(); }
    const float mu = red[0] * (1.f / DIM);
    __syncthreads();
    float s2 = 0.f;
#pragma unroll
    for (int i = 0; i < 3; i++) { float d = v[i] - mu; s2 += d * d; }
    red[t] = s2; __syncthreads();
    for (int o = 128; o > 0; o >>= 1) { if (t < o) red[t] += red[t + o]; __syncthreads(); }
    const float rstd = rsqrtf(red[0] * (1.f / DIM) + 1e-5f);
#pragma unroll
    for (int i = 0; i < 3; i++) {
        int c = t + 256 * i;
        out[(size_t)row * DIM + c] = (bf16)((v[i] - mu) * rstd * w[c] + b[c]);
    }
}

#if HAVE_TDM
// ---------------- TDM: DMA a 64-row x 32-elem bf16 tile (row stride = strideElems) to LDS ----------
// D# per CDNA5 ISA ch.8: group0 {count=1, lds_addr, global_addr, type=2},
// group1 {data_size=2B, tensor_dim0=32, tensor_dim1=64, tile_dim0=32, tile_dim1=64, dim0_stride}.
__device__ __forceinline__ void tdm_load_64x32(const bf16* gsrc, unsigned ldsOff, int strideElems) {
    const unsigned long long ga = (unsigned long long)(size_t)gsrc;
    u32x4 g0;
    g0[0] = 1u;                                                 // count=1, no gather
    g0[1] = ldsOff;                                             // lds_addr
    g0[2] = (unsigned)ga;                                       // global_addr[31:0]
    g0[3] = ((unsigned)(ga >> 32) & 0x01FFFFFFu) | 0x80000000u; // addr[56:32] | type=2
    i32x8 g1 = {};
    g1[0] = (int)0x00010000;       // workgroup_mask=0, data_size=1 (2 bytes)
    g1[1] = (int)(32u << 16);      // tensor_dim0 = 32 (lo16 at bits 79:64... packed [31:16])
    g1[2] = (int)(64u << 16);      // tensor_dim1 = 64
    g1[3] = (int)(32u << 16);      // tile_dim0 = 32
    g1[4] = (int)64;               // tile_dim1 = 64, tile_dim2 = 0
    g1[5] = strideElems;           // tensor_dim0_stride (data_size units)
    i32x4 g2 = {}, g3 = {};
#if __clang_major__ >= 23
    i32x8 g4 = {};
    __builtin_amdgcn_tensor_load_to_lds(g0, g1, g2, g3, g4, 0);
#else
    __builtin_amdgcn_tensor_load_to_lds(g0, g1, g2, g3, 0);
#endif
}
#endif

// ---------------- bf16 WMMA GEMM:  C[M,Nout] = A[M,K] * W[Nout,K]^T + bias ----------------
// Block: 8 waves, each wave computes 32 rows x 64 cols (2x4 accumulators).
// B tile staged to LDS by TDM (double-buffered, overlapped on TENSORcnt) when available.
// EPI: 0 = bf16 out + q-scale on first DIM cols, 1 = bf16 out + exact GELU, 2 = f32 out + residual
template<int EPI>
__global__ __launch_bounds__(256) void gemm_bf16_kernel(
    const bf16*  __restrict__ A, const bf16* __restrict__ W,
    const float* __restrict__ bias, const float* __restrict__ resid,
    bf16* __restrict__ outb, float* __restrict__ outf,
    int M, int K, int Nout)
{
    __shared__ __align__(32) bf16 sB[2][64 * 32];        // [buf][n within tile][k within tile]
    const int lane = threadIdx.x & 31, wave = threadIdx.x >> 5;
    const int m0 = blockIdx.x * 256 + wave * 32;
    const int n0 = blockIdx.y * 64;
    const int lrow = lane & 15;
    const int hoff = lane & 16;                          // 0 / 16 (B-fragment K offset)
    const int aoff = hoff;                               // 0 / 16 (A-fragment K offset)
    int rowc[2];
    rowc[0] = min(m0 + lrow, M - 1);
    rowc[1] = min(m0 + 16 + lrow, M - 1);

    v8f acc[2][4] = {};
#if HAVE_TDM
    const bool issuer = (threadIdx.x < 32);              // wave 0 drives the TDM
    if (issuer) tdm_load_64x32(&W[(size_t)n0 * K], (unsigned)(size_t)&sB[0][0], K);
#endif
    for (int kb = 0; kb < K; kb += 32) {
        const int buf = (kb >> 5) & 1;
#if HAVE_TDM
        if (issuer) {
            if (kb + 32 < K) {                           // prefetch next tile via DMA
                tdm_load_64x32(&W[(size_t)n0 * K + kb + 32],
                               (unsigned)(size_t)&sB[buf ^ 1][0], K);
                __builtin_amdgcn_s_wait_tensorcnt(1);    // oldest (current buf) complete
            } else {
                __builtin_amdgcn_s_wait_tensorcnt(0);
            }
        }
        __syncthreads();
#else
        {
            const int n = threadIdx.x >> 2, kc = (threadIdx.x & 3) * 8;
            *(v8bf*)&sB[buf][n * 32 + kc] = *(const v8bf*)&W[(size_t)(n0 + n) * K + kb + kc];
        }
        __syncthreads();
#endif
        // A fragments (16x32 bf16, k = 16*half + e), two row sub-tiles
        v16bf a[2];
#pragma unroll
        for (int mt = 0; mt < 2; mt++) {
            const bf16* ap = &A[(size_t)rowc[mt] * K + kb + aoff];
            v8bf lo = *(const v8bf*)ap;
            v8bf hi = *(const v8bf*)(ap + 8);
#pragma unroll
            for (int e = 0; e < 8; e++) { a[mt][e] = lo[e]; a[mt][e + 8] = hi[e]; }
        }
#pragma unroll
        for (int nt = 0; nt < 4; nt++) {
            v16bf bm = *(const v16bf*)&sB[buf][(nt * 16 + lrow) * 32 + hoff];
#pragma unroll
            for (int mt = 0; mt < 2; mt++)
                acc[mt][nt] = __builtin_amdgcn_wmma_f32_16x16x32_bf16(
                    false, a[mt], false, bm, (short)0, acc[mt][nt], false, false);
        }
        __syncthreads();
    }
    // epilogue
    const int half = lane >> 4;
#pragma unroll
    for (int nt = 0; nt < 4; nt++) {
        const int col = n0 + nt * 16 + lrow;
        const float bc = bias[col];
#pragma unroll
        for (int mt = 0; mt < 2; mt++)
#pragma unroll
            for (int r = 0; r < 8; r++) {
                const int rr = m0 + mt * 16 + r + 8 * half;
                if (rr < M) {
                    float v = acc[mt][nt][r] + bc;
                    const size_t idx = (size_t)rr * Nout + col;
                    if (EPI == 0 && col < DIM) v *= 0.125f;   // q * head_dim^-0.5
                    if (EPI == 1) v = 0.5f * v * (1.0f + erff(v * 0.70710678118654752f));
                    if (EPI == 2) outf[idx] = v + resid[idx];
                    else          outb[idx] = (bf16)v;
                }
            }
    }
}

// ---------------- flash attention, fully transposed (keys = C-rows, queries = C-cols) ----------------
__global__ __launch_bounds__(128) void attn_kernel(
    const bf16* __restrict__ qkv, const float* __restrict__ biasS,
    bf16* __restrict__ outb)
{
    __shared__ __align__(32) bf16 sP[4][16 * 32];        // per-wave P: [query][key]
    __shared__ __align__(32) bf16 sV[64 * 32];           // V tile transposed: [d][tok]
    const int lane = threadIdx.x & 31, wave = threadIdx.x >> 5;
    const int b = blockIdx.z, h = blockIdx.y;
    const int qt = blockIdx.x * 4 + wave;
    const int q0 = qt * 16;
    const int lrow = lane & 15, half = lane >> 4;
    const int hoff = lane & 16, aoff = hoff;
    const bf16* bbase = qkv + (size_t)(b * N_TOK) * QKV3;

    // Q as B-fragments (64 x 16): col = query = lane&15, element e -> kdim = ks*32 + e + hoff
    v16bf bQ[2];
    {
        const int qrow = min(q0 + lrow, N_TOK - 1);
        const bf16* qb = bbase + (size_t)qrow * QKV3 + h * HD;
#pragma unroll
        for (int ks = 0; ks < 2; ks++)
            bQ[ks] = *(const v16bf*)(qb + ks * 32 + hoff);
    }
    const float* bp = biasS + ((size_t)h * (NQT * NKT) + (size_t)min(qt, NQT - 1) * NKT) * 512
                    + lane * 16;

    float mprev = -1.0e30f, lsum = 0.f;                  // per-query scalars (query = lane&15)
    v8f accOT[4] = {};                                   // O^T: rows d = dt*16 + r + 8*half

    auto do_tile = [&](int t0, bool clampTok) {
        {
            const int tok = threadIdx.x >> 2, dc = (threadIdx.x & 3) * 16;
            const int tg = clampTok ? min(t0 + tok, N_TOK - 1) : (t0 + tok);
            v16bf vv = *(const v16bf*)(bbase + (size_t)tg * QKV3 + 2 * DIM + h * HD + dc);
#pragma unroll
            for (int e = 0; e < 16; e++) sV[(dc + e) * 32 + tok] = vv[e];
        }
        __syncthreads();

        // S' = K * Q^T : two 16x16 tiles along keys (A = K-subtile, k = ks*32 + 16*half + e)
        v8f accT[2] = {};
#pragma unroll
        for (int hs = 0; hs < 2; hs++) {
            const int tok = clampTok ? min(t0 + hs * 16 + lrow, N_TOK - 1)
                                     : (t0 + hs * 16 + lrow);
            const bf16* kb = bbase + (size_t)tok * QKV3 + DIM + h * HD;
#pragma unroll
            for (int ks = 0; ks < 2; ks++) {
                v16bf aK;
                v8bf lo = *(const v8bf*)(kb + ks * 32 + aoff);
                v8bf hi = *(const v8bf*)(kb + ks * 32 + aoff + 8);
#pragma unroll
                for (int e = 0; e < 8; e++) { aK[e] = lo[e]; aK[e + 8] = hi[e]; }
                accT[hs] = __builtin_amdgcn_wmma_f32_16x16x32_bf16(
                    false, aK, false, bQ[ks], (short)0, accT[hs], false, false);
            }
        }

        // add pre-swizzled bias (two contiguous v8f per lane; key mask baked in)
        {
            v8f b0 = *(const v8f*)(bp);
            v8f b1 = *(const v8f*)(bp + 8);
#pragma unroll
            for (int r = 0; r < 8; r++) { accT[0][r] += b0[r]; accT[1][r] += b1[r]; }
            bp += 512;
        }

        // online softmax: reduction over keys is in-lane + one xor16
        float mt = fmaxf(accT[0][0], accT[1][0]);
#pragma unroll
        for (int r = 1; r < 8; r++) mt = fmaxf(mt, fmaxf(accT[0][r], accT[1][r]));
        mt = fmaxf(mt, __shfl_xor(mt, 16, 32));
        const float mn = fmaxf(mprev, mt);
        const float corr = __expf(mprev - mn);
        mprev = mn;
        float ps = 0.f;
#pragma unroll
        for (int hs = 0; hs < 2; hs++)
#pragma unroll
            for (int r = 0; r < 8; r++) {
                const float p = __expf(accT[hs][r] - mn);
                accT[hs][r] = p;
                ps += p;
            }
        ps += __shfl_xor(ps, 16, 32);
        lsum = lsum * corr + ps;
#pragma unroll
        for (int dt = 0; dt < 4; dt++)
#pragma unroll
            for (int r = 0; r < 8; r++) accOT[dt][r] *= corr;

        // P -> LDS [query][key]: two packed 16B stores per lane
#pragma unroll
        for (int hs = 0; hs < 2; hs++) {
            v8bf pk;
#pragma unroll
            for (int r = 0; r < 8; r++) pk[r] = (bf16)accT[hs][r];
            *(v8bf*)&sP[wave][lrow * 32 + hs * 16 + 8 * half] = pk;
        }
        __syncthreads();
        // P as B-fragment: col = query = lane&15, element e -> key = e + hoff (one 32B load)
        v16bf bP = *(const v16bf*)&sP[wave][lrow * 32 + hoff];
        // O^T += V^T * P : A = V^T sub-tiles (16 d-rows x 32 keys) from sV
#pragma unroll
        for (int dt = 0; dt < 4; dt++) {
            v16bf aV;
            const bf16* vp = &sV[(dt * 16 + lrow) * 32 + aoff];
            v8bf lo = *(const v8bf*)vp;
            v8bf hi = *(const v8bf*)(vp + 8);
#pragma unroll
            for (int e = 0; e < 8; e++) { aV[e] = lo[e]; aV[e + 8] = hi[e]; }
            accOT[dt] = __builtin_amdgcn_wmma_f32_16x16x32_bf16(
                false, aV, false, bP, (short)0, accOT[dt], false, false);
        }
        __syncthreads();
    };

    for (int kt = 0; kt < NKT - 1; kt++) do_tile(kt * 32, false);   // clamp-free hot loop
    do_tile((NKT - 1) * 32, true);                                  // peeled tail tile

    // normalize + store: per lane one query column, 4 packed 16B stores
    const int qg = q0 + lrow;
    if (qg < N_TOK) {
        const float inv = 1.0f / lsum;
        bf16* ob = outb + ((size_t)(b * N_TOK) + qg) * DIM + h * HD;
#pragma unroll
        for (int dt = 0; dt < 4; dt++) {
            v8bf po;
#pragma unroll
            for (int r = 0; r < 8; r++) po[r] = (bf16)(accOT[dt][r] * inv);
            *(v8bf*)(ob + dt * 16 + 8 * half) = po;
        }
    }
}

// ---------------- host orchestration ----------------
extern "C" void kernel_launch(void* const* d_in, const int* in_sizes, int n_in,
                              void* d_out, int out_size, void* d_ws, size_t ws_size,
                              hipStream_t stream) {
    (void)in_sizes; (void)n_in; (void)out_size; (void)ws_size;
    const float* x      = (const float*)d_in[0];
    const float* qkv_w  = (const float*)d_in[3];
    const float* qkv_b  = (const float*)d_in[4];
    const float* proj_w = (const float*)d_in[5];
    const float* proj_b = (const float*)d_in[6];
    const float* rel_t  = (const float*)d_in[7];
    const float* n1w    = (const float*)d_in[8];
    const float* n1b    = (const float*)d_in[9];
    const float* n2w    = (const float*)d_in[10];
    const float* n2b    = (const float*)d_in[11];
    const float* f1w    = (const float*)d_in[12];
    const float* f1b    = (const float*)d_in[13];
    const float* f2w    = (const float*)d_in[14];
    const float* f2b    = (const float*)d_in[15];
    float* out = (float*)d_out;

    char* ws = (char*)d_ws;
    const size_t OFF_WQ = 0;
    const size_t OFF_WP = OFF_WQ + (size_t)DIM * QKV3 * 2;
    const size_t OFF_W1 = OFF_WP + (size_t)DIM * DIM * 2;
    const size_t OFF_W2 = OFF_W1 + (size_t)HID * DIM * 2;
    const size_t OFF_R1 = OFF_W2 + (size_t)DIM * HID * 2;
    const size_t OFF_R2 = OFF_R1 + (size_t)M_ROWS * DIM * 2;
    const size_t OFF_BI = OFF_R2 + (size_t)M_ROWS * HID * 2;   // R2 sized for mlp hidden
    bf16*  Wq = (bf16*)(ws + OFF_WQ);
    bf16*  Wp = (bf16*)(ws + OFF_WP);
    bf16*  W1 = (bf16*)(ws + OFF_W1);
    bf16*  W2 = (bf16*)(ws + OFF_W2);
    bf16*  R1 = (bf16*)(ws + OFF_R1);   // xn1 -> attn_out -> xn2
    bf16*  R2 = (bf16*)(ws + OFF_R2);   // qkv -> mlp hidden
    float* BI = (float*)(ws + OFF_BI);  // [NH][NQT][NKT][512] fragment-order bias

    cvt_bf16_kernel<<<(DIM * QKV3 + 255) / 256, 256, 0, stream>>>(qkv_w, Wq, DIM * QKV3);
    cvt_bf16_kernel<<<(DIM * DIM  + 255) / 256, 256, 0, stream>>>(proj_w, Wp, DIM * DIM);
    cvt_bf16_kernel<<<(HID * DIM  + 255) / 256, 256, 0, stream>>>(f1w, W1, HID * DIM);
    cvt_bf16_kernel<<<(DIM * HID  + 255) / 256, 256, 0, stream>>>(f2w, W2, DIM * HID);
    bias_kernel<<<dim3(NQT * NKT, NH), 512, 0, stream>>>(rel_t, BI);

    const int gM = (M_ROWS + 255) / 256;                       // 99

    ln_bf16_kernel<<<M_ROWS, 256, 0, stream>>>(x, n1w, n1b, R1);
    gemm_bf16_kernel<0><<<dim3(gM, QKV3 / 64), 256, 0, stream>>>(
        R1, Wq, qkv_b, nullptr, R2, nullptr, M_ROWS, DIM, QKV3);
    attn_kernel<<<dim3((N_TOK + 63) / 64, NH, BATCH), 128, 0, stream>>>(R2, BI, R1);
    gemm_bf16_kernel<2><<<dim3(gM, DIM / 64), 256, 0, stream>>>(
        R1, Wp, proj_b, x, nullptr, out, M_ROWS, DIM, DIM);
    ln_bf16_kernel<<<M_ROWS, 256, 0, stream>>>(out, n2w, n2b, R1);
    gemm_bf16_kernel<1><<<dim3(gM, HID / 64), 256, 0, stream>>>(
        R1, W1, f1b, nullptr, R2, nullptr, M_ROWS, DIM, HID);
    gemm_bf16_kernel<2><<<dim3(gM, DIM / 64), 256, 0, stream>>>(
        R2, W2, f2b, out, nullptr, out, M_ROWS, HID, DIM);
}
